// VAKT_52226802319686
// MI455X (gfx1250) — matrix operand order repaired and checked by reading, not compiled, not run on hardware
//
#include <hip/hip_runtime.h>

// ============================================================================
// AKT-style transformer for MI455X (gfx1250, wave32, WMMA).
// All GEMMs run on v_wmma_f32_16x16x32_f16 (f16 inputs, f32 accumulate).
// Fragment loads are vectorized (ds_load_b128 pairs) by giving A and B the same
// contiguous k order (k = 16*hf + s); weight/V tiles are staged into LDS in fragment order.
// Softmax / distance-decay / LayerNorm stay in f32 VALU.
// Workspace assumes ws_size >= ~140 MB.
// ============================================================================

typedef _Float16 half_t;
typedef __attribute__((ext_vector_type(16))) _Float16 v16h;
typedef __attribute__((ext_vector_type(8)))  _Float16 v8h;
typedef __attribute__((ext_vector_type(8)))  float    v8f;

#define NEGV (-1.0e32f)

__device__ __forceinline__ v8f wmma_f16(v16h a, v16h b, v8f c) {
  return __builtin_amdgcn_wmma_f32_16x16x32_f16(
      false, a, false, b, (short)0, c, false, false);
}

__device__ __forceinline__ v16h cat8(v8h lo, v8h hi) {
  return __builtin_shufflevector(lo, hi, 0, 1, 2, 3, 4, 5, 6, 7,
                                 8, 9, 10, 11, 12, 13, 14, 15);
}

// A fragment (16x32 f16) from a natural row-major tile. The lane's 16
// elements are one contiguous 16-half run: [16*hf, +16).
// 'stride' (elements) must keep rows 16-byte aligned.
__device__ __forceinline__ v16h load_a_frag_v(const half_t* base, int stride, int lane) {
  int row = lane & 15, hf = lane >> 4;
  const half_t* p = base + row * stride + 16 * hf;
  return cat8(*(const v8h*)p, *(const v8h*)(p + 8));
}

// B fragment where the source is [16 n-rows][K] row-major (transposed use,
// Q*K^T): lane n reads contiguous kk in [16*hf, 16*hf+16).
__device__ __forceinline__ v16h load_bT_frag_v(const half_t* base, int stride, int lane) {
  int n = lane & 15, hf = lane >> 4;
  const half_t* p = base + n * stride + 16 * hf;
  return cat8(*(const v8h*)p, *(const v8h*)(p + 8));
}

// B fragment from a fragment-ordered LDS sub-tile (512 halfs per 32x16 tile):
// element (kk,nn) stored at (nn*2 + (kk>>4))*16 + (kk&15)  -> lane reads 16
// contiguous halfs.
__device__ __forceinline__ v16h load_b_fragorder(const half_t* base, int lane) {
  int n = lane & 15, hf = lane >> 4;
  const half_t* p = base + (n * 2 + hf) * 16;
  return cat8(*(const v8h*)p, *(const v8h*)(p + 8));
}

// ---------------------------------------------------------------------------
// f32 -> f16 conversion / padding
// ---------------------------------------------------------------------------
__global__ void k_f32h(const float* __restrict__ src, half_t* __restrict__ dst, long n) {
  long i = (long)blockIdx.x * blockDim.x + threadIdx.x;
  if (i < n) dst[i] = (half_t)src[i];
}

__global__ void k_padw(const float* __restrict__ src, half_t* __restrict__ dst,
                       int K, int N, int Np) {
  long i = (long)blockIdx.x * blockDim.x + threadIdx.x;
  if (i >= (long)K * Np) return;
  int k = (int)(i / Np), n = (int)(i % Np);
  dst[i] = (n < N) ? (half_t)src[(long)k * N + n] : (half_t)0.0f;
}

// ---------------------------------------------------------------------------
// Embedding:  x0 = c_emb[c];  y = ca_emb[(ca-c)/1000] + x0
// ---------------------------------------------------------------------------
__global__ void k_embed(const int* __restrict__ c, const int* __restrict__ ca,
                        const float* __restrict__ cemb, const float* __restrict__ caemb,
                        float* __restrict__ x0F, half_t* __restrict__ x0H,
                        float* __restrict__ yF, half_t* __restrict__ yH) {
  long bs = blockIdx.x;
  int idx  = c[bs];
  int resp = (ca[bs] - idx) / 1000;
  for (int d = threadIdx.x; d < 512; d += 256) {
    float x = cemb[(long)idx * 512 + d];
    float y = caemb[(long)resp * 512 + d] + x;
    x0F[bs * 512 + d] = x;  x0H[bs * 512 + d] = (half_t)x;
    yF[bs * 512 + d]  = y;  yH[bs * 512 + d]  = (half_t)y;
  }
}

// ---------------------------------------------------------------------------
// WMMA GEMM:  C[M,Nout] = act(A[M,K] * W[K,Np] + bias)
// 256 threads = 8 waves; block tile 64(M) x 128(N); K stepped by 32.
// Each wave: 2x2 accumulators (32x32) -> 4 WMMAs per K-step with frag reuse.
// ---------------------------------------------------------------------------
__global__ void k_gemm(const half_t* __restrict__ A, const half_t* __restrict__ W,
                       const float* __restrict__ bias,
                       float* Cf, half_t* Ch,
                       int M, int K, int Np, int Nout, int act) {
  __shared__ __align__(16) half_t As[64][40];    // natural layout (+pad, rows 80B)
  __shared__ __align__(16) half_t Bs[8 * 512];   // 8 frag-ordered 32x16 sub-tiles
  int tid = threadIdx.x, lane = tid & 31, wave = tid >> 5;
  int wm2 = wave & 1, wn2 = wave >> 1;           // 2 x 4 wave grid
  long m0 = (long)blockIdx.y * 64;
  long n0 = (long)blockIdx.x * 128;

  v8f acc00 = {}, acc01 = {}, acc10 = {}, acc11 = {};

  for (int k0 = 0; k0 < K; k0 += 32) {
    if (lane == 0 && k0 + 32 < K)
      __builtin_prefetch(&W[(long)(k0 + 32) * Np + n0], 0, 1);

    // stage A: 64x32, one v8h per thread
    {
      int r = tid >> 2, c8 = (tid & 3) * 8;
      *(v8h*)&As[r][c8] = *(const v8h*)&A[(m0 + r) * K + k0 + c8];
    }
    // stage B frag-ordered: 32x128, two v8h chunks per thread
#pragma unroll
    for (int c = 0; c < 2; ++c) {
      int ci = tid + 256 * c;                    // 0..511
      int kk = ci >> 4, n8 = (ci & 15) * 8;
      v8h w = *(const v8h*)&W[(long)(k0 + kk) * Np + n0 + n8];
#pragma unroll
      for (int e = 0; e < 8; ++e) {
        int n = n8 + e;
        Bs[(n >> 4) * 512 + ((n & 15) * 2 + (kk >> 4)) * 16 + (kk & 15)] = w[e];
      }
    }
    __syncthreads();

    v16h a0 = load_a_frag_v(&As[wm2 * 32][0], 40, lane);
    v16h a1 = load_a_frag_v(&As[wm2 * 32 + 16][0], 40, lane);
    v16h b0 = load_b_fragorder(Bs + (wn2 * 2 + 0) * 512, lane);
    v16h b1 = load_b_fragorder(Bs + (wn2 * 2 + 1) * 512, lane);
    acc00 = wmma_f16(a0, b0, acc00);
    acc01 = wmma_f16(a0, b1, acc01);
    acc10 = wmma_f16(a1, b0, acc10);
    acc11 = wmma_f16(a1, b1, acc11);
    __syncthreads();
  }

  int hf = lane >> 4, n = lane & 15;
  v8f accs[2][2] = {{acc00, acc01}, {acc10, acc11}};
#pragma unroll
  for (int mi = 0; mi < 2; ++mi)
#pragma unroll
    for (int ni = 0; ni < 2; ++ni) {
      long col = n0 + wn2 * 32 + ni * 16 + n;
      if (col < Nout) {
        float bv = bias ? bias[col] : 0.0f;
#pragma unroll
        for (int r = 0; r < 8; ++r) {
          long row = m0 + wm2 * 32 + mi * 16 + r + 8 * hf;
          float v = accs[mi][ni][r] + bv;
          if (act) v = fmaxf(v, 0.0f);
          if (Cf) Cf[row * Nout + col] = v;
          if (Ch) Ch[row * Nout + col] = (half_t)v;
        }
      }
    }
}

// ---------------------------------------------------------------------------
// Fused distance-decay attention for one (b, h, 16-q-row tile).
// maskType: 0 -> k<=q (tril), 1 -> k<q (tril,-1).  gamma = -softplus(gammas[h])
// ---------------------------------------------------------------------------
__global__ void k_attn(const half_t* __restrict__ Q, const half_t* __restrict__ Km,
                       const half_t* __restrict__ V, const float* __restrict__ gammas,
                       int maskType, int zeroPad, half_t* __restrict__ O) {
  const int S = 512, D = 512, DK = 64;
  int qt = blockIdx.x, h = blockIdx.y, b = blockIdx.z;
  int q0 = qt * 16;
  int tid = threadIdx.x, lane = tid & 31, wave = tid >> 5;

  __shared__ __align__(16) half_t Qs[16][72];       // q tile 16x64 (rows 144B)
  __shared__ __align__(16) half_t Kst[8][16][72];   // per-wave K staging
  __shared__ __align__(16) half_t Ah[16][512];      // attn probs (f16)
  __shared__ __align__(16) half_t Vt[4 * 512];      // frag-ordered V tile 32x64
  __shared__ float  Sc[16][512];                    // scores
  __shared__ float  Pc[16][512];                    // p / cumsum
  __shared__ float  red[16][17];
  __shared__ float  rstat[16][4];

  float gamma = -log1pf(expf(gammas[h]));           // -softplus

  // stage Q tile (128 v8h chunks)
  if (tid < 128) {
    int r = tid >> 3, c8 = (tid & 7) * 8;
    *(v8h*)&Qs[r][c8] = *(const v8h*)&Q[((long)(b * S + q0 + r)) * D + h * DK + c8];
  }
  __syncthreads();

  // ---- phase 1: scores = Q K^T / sqrt(dk); each wave does 4 key tiles ----
  for (int t = 0; t < 4; ++t) {
    int kt = t * 8 + wave;
#pragma unroll
    for (int vi = lane; vi < 128; vi += 32) {       // 16x64 tile, v8h chunks
      int r = vi >> 3, c8 = (vi & 7) * 8;
      *(v8h*)&Kst[wave][r][c8] =
          *(const v8h*)&Km[((long)(b * S + kt * 16 + r)) * D + h * DK + c8];
    }
    __syncthreads();
    v8f acc = {};
#pragma unroll
    for (int kc = 0; kc < 2; ++kc) {
      v16h a  = load_a_frag_v(&Qs[0][kc * 32], 72, lane);
      v16h bb = load_bT_frag_v(&Kst[wave][0][kc * 32], 72, lane);
      acc = wmma_f16(a, bb, acc);
    }
    int hf = lane >> 4, n = lane & 15;
#pragma unroll
    for (int r = 0; r < 8; ++r)
      Sc[r + 8 * hf][kt * 16 + n] = acc[r] * 0.125f;   // 1/sqrt(64)
    __syncthreads();
  }

  // ---- phase 2: decay pipeline, 16 threads per q row ----
  int row = tid >> 4, tc = tid & 15;
  int q = q0 + row;

  float mx = -3.0e38f;
  for (int k = tc; k < S; k += 16) {
    bool m = maskType == 0 ? (k <= q) : (k < q);
    if (m) mx = fmaxf(mx, Sc[row][k]);
  }
  red[row][tc] = mx; __syncthreads();
  if (tc == 0) { float v = red[row][0];
    for (int i = 1; i < 16; ++i) v = fmaxf(v, red[row][i]); rstat[row][0] = v; }
  __syncthreads();
  float rmax = rstat[row][0];

  float psum = 0.0f;
  for (int k = tc; k < S; k += 16) {
    bool m = maskType == 0 ? (k <= q) : (k < q);
    float e = m ? expf(Sc[row][k] - rmax) : 0.0f;
    Pc[row][k] = e; psum += e;
  }
  red[row][tc] = psum; __syncthreads();
  if (tc == 0) { float v = 0.0f;
    for (int i = 0; i < 16; ++i) v += red[row][i]; rstat[row][1] = v; }
  __syncthreads();
  float rsum = rstat[row][1];
  float inv = rsum > 0.0f ? 1.0f / rsum : 0.0f;
  for (int k = tc; k < S; k += 16) Pc[row][k] *= inv;
  __syncthreads();

  if (tc == 0) {                                    // serial cumsum per row
    float c = 0.0f;
    for (int k = 0; k < S; ++k) { c += Pc[row][k]; Pc[row][k] = c; }
    rstat[row][2] = c;                              // disttotal
  }
  __syncthreads();
  float total = rstat[row][2];

  for (int k = tc; k < S; k += 16) {
    bool m = maskType == 0 ? (k <= q) : (k < q);
    float s2;
    if (m) {
      float d    = total - Pc[row][k];
      float pos  = fabsf((float)(q - k));
      float dist = sqrtf(fmaxf(d * pos, 0.0f));
      float eff  = fminf(fmaxf(expf(dist * gamma), 1e-5f), 1e5f);
      s2 = Sc[row][k] * eff;
    } else s2 = NEGV;
    Sc[row][k] = s2;
  }
  __syncthreads();

  // second softmax over full row (all-NEG row -> uniform, matches reference;
  // zero_pad then clears q==0)
  mx = -3.0e38f;
  for (int k = tc; k < S; k += 16) mx = fmaxf(mx, Sc[row][k]);
  red[row][tc] = mx; __syncthreads();
  if (tc == 0) { float v = red[row][0];
    for (int i = 1; i < 16; ++i) v = fmaxf(v, red[row][i]); rstat[row][3] = v; }
  __syncthreads();
  float rmax2 = rstat[row][3];
  float asum = 0.0f;
  for (int k = tc; k < S; k += 16) {
    float e = expf(Sc[row][k] - rmax2);
    Pc[row][k] = e; asum += e;
  }
  red[row][tc] = asum; __syncthreads();
  if (tc == 0) { float v = 0.0f;
    for (int i = 0; i < 16; ++i) v += red[row][i]; rstat[row][1] = v; }
  __syncthreads();
  float inv2 = 1.0f / rstat[row][1];
  bool killRow = (zeroPad != 0) && (q == 0);
  for (int k = tc; k < S; k += 16)
    Ah[row][k] = (half_t)(killRow ? 0.0f : Pc[row][k] * inv2);
  __syncthreads();

  // ---- phase 3: out(16x64) = attn(16x512) @ V(512x64) ----
  // Each K-step: all 256 threads stage V tile (32 keys x 64 dims) into
  // frag-ordered LDS with v8h global loads; waves 0..3 each own one n-tile.
  v8f acc = {};
  int hf = lane >> 4, n = lane & 15;
  for (int kt = 0; kt < 16; ++kt) {
    {
      int kk = tid >> 3, d8 = (tid & 7) * 8;        // 256 chunks = 32x64
      v8h w = *(const v8h*)&V[((long)(b * S + kt * 32 + kk)) * D + h * DK + d8];
#pragma unroll
      for (int e = 0; e < 8; ++e) {
        int d = d8 + e;
        Vt[(d >> 4) * 512 + ((d & 15) * 2 + (kk >> 4)) * 16 + (kk & 15)] = w[e];
      }
    }
    __syncthreads();
    if (wave < 4) {
      v16h a  = load_a_frag_v(&Ah[0][kt * 32], 512, lane);
      v16h bb = load_b_fragorder(Vt + wave * 512, lane);
      acc = wmma_f16(a, bb, acc);
    }
    __syncthreads();
  }
  if (wave < 4) {
#pragma unroll
    for (int r = 0; r < 8; ++r) {
      int m = r + 8 * hf;
      O[((long)(b * S + q0 + m)) * D + h * DK + wave * 16 + n] = (half_t)acc[r];
    }
  }
}

// ---------------------------------------------------------------------------
// Residual add + LayerNorm over D=512 (block per row). out may alias A.
// ---------------------------------------------------------------------------
__global__ void k_add_ln(const float* __restrict__ A, const float* __restrict__ Bv,
                         const float* __restrict__ s, const float* __restrict__ bb,
                         float* outF, half_t* outH) {
  long row = blockIdx.x;
  int tid = threadIdx.x;
  __shared__ float red[256];
  long base = row * 512;
  float v0 = A[base + tid] + Bv[base + tid];
  float v1 = A[base + tid + 256] + Bv[base + tid + 256];

  red[tid] = v0 + v1; __syncthreads();
  for (int st = 128; st > 0; st >>= 1) {
    if (tid < st) red[tid] += red[tid + st];
    __syncthreads();
  }
  float mean = red[0] * (1.0f / 512.0f); __syncthreads();

  float d0 = v0 - mean, d1 = v1 - mean;
  red[tid] = d0 * d0 + d1 * d1; __syncthreads();
  for (int st = 128; st > 0; st >>= 1) {
    if (tid < st) red[tid] += red[tid + st];
    __syncthreads();
  }
  float rstd = rsqrtf(red[0] * (1.0f / 512.0f) + 1e-5f);

  float o0 = d0 * rstd * s[tid] + bb[tid];
  float o1 = d1 * rstd * s[tid + 256] + bb[tid + 256];
  if (outF) { outF[base + tid] = o0; outF[base + tid + 256] = o1; }
  if (outH) { outH[base + tid] = (half_t)o0; outH[base + tid + 256] = (half_t)o1; }
}

// ---------------------------------------------------------------------------
// hcat = [x, x0]  (f16)
// ---------------------------------------------------------------------------
__global__ void k_concat(const half_t* __restrict__ xH, const half_t* __restrict__ x0H,
                         half_t* __restrict__ hc) {
  long row = blockIdx.x;
  for (int c = threadIdx.x; c < 1024; c += 256)
    hc[row * 1024 + c] = (c < 512) ? xH[row * 512 + c] : x0H[row * 512 + (c - 512)];
}

// ===========================================================================
// Host orchestration
// ===========================================================================
extern "C" void kernel_launch(void* const* d_in, const int* in_sizes, int n_in,
                              void* d_out, int out_size, void* d_ws, size_t ws_size,
                              hipStream_t stream) {
  (void)in_sizes; (void)n_in; (void)out_size; (void)ws_size;
  const int  B = 8, S = 512, D = 512, FF = 2048, NH = 8, Lx = 6;
  const long MS = (long)B * S;

  const int*   c_data  = (const int*)  d_in[0];
  const int*   ca_data = (const int*)  d_in[1];
  const float* c_emb   = (const float*)d_in[2];
  const float* ca_emb  = (const float*)d_in[3];
  const float* WkF  = (const float*)d_in[4];
  const float* bkF  = (const float*)d_in[5];
  const float* WvF  = (const float*)d_in[6];
  const float* bvF  = (const float*)d_in[7];
  const float* WoF  = (const float*)d_in[8];
  const float* boF  = (const float*)d_in[9];
  const float* gF   = (const float*)d_in[10];
  const float* ln1s = (const float*)d_in[11];
  const float* ln1b = (const float*)d_in[12];
  const float* W1F  = (const float*)d_in[13];
  const float* b1F  = (const float*)d_in[14];
  const float* W2F  = (const float*)d_in[15];
  const float* b2F  = (const float*)d_in[16];
  const float* ln2s = (const float*)d_in[17];
  const float* ln2b = (const float*)d_in[18];
  const float* oW0F = (const float*)d_in[19];
  const float* ob0  = (const float*)d_in[20];
  const float* oW1F = (const float*)d_in[21];
  const float* ob1  = (const float*)d_in[22];
  const float* oW2F = (const float*)d_in[23];
  const float* ob2  = (const float*)d_in[24];
  float* out = (float*)d_out;

  char* base = (char*)d_ws; size_t off = 0;
  auto alloc = [&](size_t bytes) -> void* {
    void* p = base + off; off = (off + bytes + 255) & ~(size_t)255; return p;
  };
  float*  x0F = (float*)alloc(MS * D * 4);
  float*  yF  = (float*)alloc(MS * D * 4);
  float*  xF  = (float*)alloc(MS * D * 4);
  float*  t0F = (float*)alloc(MS * D * 4);
  float*  t1F = (float*)alloc(MS * D * 4);
  half_t* x0H = (half_t*)alloc(MS * D * 2);
  half_t* yH  = (half_t*)alloc(MS * D * 2);
  half_t* xH  = (half_t*)alloc(MS * D * 2);
  half_t* qHb = (half_t*)alloc(MS * D * 2);
  half_t* pQ  = (half_t*)alloc(MS * D * 2);
  half_t* pK  = (half_t*)alloc(MS * D * 2);
  half_t* pV  = (half_t*)alloc(MS * D * 2);
  half_t* aoH = (half_t*)alloc(MS * D * 2);
  half_t* ffH = (half_t*)alloc(MS * FF * 2);
  half_t* WkH = (half_t*)alloc((long)Lx * D * D * 2);
  half_t* WvH = (half_t*)alloc((long)Lx * D * D * 2);
  half_t* WoH = (half_t*)alloc((long)Lx * D * D * 2);
  half_t* W1H = (half_t*)alloc((long)Lx * D * FF * 2);
  half_t* W2H = (half_t*)alloc((long)Lx * FF * D * 2);
  half_t* oW0H = (half_t*)alloc(1024L * 512 * 2);
  half_t* oW1H = (half_t*)alloc(512L * 1024 * 2);
  half_t* oW2H = (half_t*)alloc(1024L * 1024 * 2);   // N padded 1000 -> 1024
  half_t* hcatH = ffH;                               // [4096,1024]
  half_t* h1H   = ffH + MS * 1024;                   // [4096,1024]
  half_t* h0H   = aoH;                               // [4096,512]

  auto cvt = [&](const float* s_, half_t* d_, long n) {
    k_f32h<<<(int)((n + 255) / 256), 256, 0, stream>>>(s_, d_, n);
  };
  cvt(WkF, WkH, (long)Lx * D * D);
  cvt(WvF, WvH, (long)Lx * D * D);
  cvt(WoF, WoH, (long)Lx * D * D);
  cvt(W1F, W1H, (long)Lx * D * FF);
  cvt(W2F, W2H, (long)Lx * FF * D);
  cvt(oW0F, oW0H, 1024L * 512);
  cvt(oW1F, oW1H, 512L * 1024);
  k_padw<<<(int)((1024L * 1024 + 255) / 256), 256, 0, stream>>>(oW2F, oW2H, 1024, 1000, 1024);

  k_embed<<<(int)MS, 256, 0, stream>>>(c_data, ca_data, c_emb, ca_emb, x0F, x0H, yF, yH);
  hipMemcpyAsync(xF, x0F, MS * D * 4, hipMemcpyDeviceToDevice, stream);
  hipMemcpyAsync(xH, x0H, MS * D * 2, hipMemcpyDeviceToDevice, stream);

  auto gemm = [&](const half_t* A, const half_t* W, const float* bias,
                  float* Cf, half_t* Ch, int M, int K, int Np, int Nout, int act) {
    dim3 grid(Np / 128, M / 64);
    k_gemm<<<grid, 256, 0, stream>>>(A, W, bias, Cf, Ch, M, K, Np, Nout, act);
  };

  auto layer = [&](int i, int maskType, float* qF, const half_t* qHin,
                   const half_t* kHin, const half_t* vHin,
                   float* outF, half_t* outH, int zeroPad, int applyPos) {
    long wq = (long)i * D * D;
    gemm(qHin, WkH + wq, bkF + i * D, nullptr, pQ, (int)MS, D, D, D, 0);   // kq_same
    gemm(kHin, WkH + wq, bkF + i * D, nullptr, pK, (int)MS, D, D, D, 0);
    gemm(vHin, WvH + wq, bvF + i * D, nullptr, pV, (int)MS, D, D, D, 0);
    k_attn<<<dim3(S / 16, NH, B), 256, 0, stream>>>(pQ, pK, pV, gF + i * NH,
                                                    maskType, zeroPad, aoH);
    gemm(aoH, WoH + wq, boF + i * D, t0F, nullptr, (int)MS, D, D, D, 0);
    if (applyPos) {
      k_add_ln<<<(int)MS, 256, 0, stream>>>(qF, t0F, ln1s + i * D, ln1b + i * D, t1F, qHb);
      gemm(qHb, W1H + (long)i * D * FF, b1F + i * FF, nullptr, ffH, (int)MS, D, FF, FF, 1);
      gemm(ffH, W2H + (long)i * FF * D, b2F + i * D, t0F, nullptr, (int)MS, FF, D, D, 0);
      k_add_ln<<<(int)MS, 256, 0, stream>>>(t1F, t0F, ln2s + i * D, ln2b + i * D, outF, outH);
    } else {
      k_add_ln<<<(int)MS, 256, 0, stream>>>(qF, t0F, ln1s + i * D, ln1b + i * D, outF, outH);
    }
  };

  for (int i = 0; i < 2; ++i)
    layer(i, /*mask1*/0, yF, yH, x0H, yH, yF, yH, 0, 1);

  for (int j = 0; j < 4; ++j) {
    int i = 2 + j;
    if ((j & 1) == 0)
      layer(i, /*mask1*/0, xF, xH, xH, xH, xF, xH, 0, 0);
    else
      layer(i, /*mask0*/1, xF, xH, xH, yH, xF, xH, 1, 1);
  }

  k_concat<<<(int)MS, 256, 0, stream>>>(xH, x0H, hcatH);
  gemm(hcatH, oW0H, ob0, nullptr, h0H, (int)MS, 1024, 512, 512, 1);
  gemm(h0H,   oW1H, ob1, nullptr, h1H, (int)MS, 512, 1024, 1024, 1);
  gemm(h1H,   oW2H, ob2, out, nullptr, (int)MS, 1024, 1024, 1000, 0);
}
